// BaseAudioQuantizer_72499047957277
// MI455X (gfx1250) — matrix operand order repaired and checked by reading, not compiled, not run on hardware
//
#include <hip/hip_runtime.h>

// ---------------------------------------------------------------------------
// VQ codebook assignment for MI455X (gfx1250, wave32, WMMA).
//   argmin ranking:  score = |c|^2 - 2 x.c     (|x|^2 is row-constant)
//   x.c computed with bf16x3 split precision on V_WMMA_F32_16X16X32_BF16:
//       x = xh + xl, c = ch + cl (bf16);  x.c ~= xh.ch + xh.cl + xl.ch
//   -> ~2.7x fewer matrix issues than fp32 16x16x4 WMMA at ~2^-15 rel error.
// ---------------------------------------------------------------------------

typedef __bf16 bf16;
typedef bf16  v16bf __attribute__((ext_vector_type(16)));
typedef float v8f   __attribute__((ext_vector_type(8)));
typedef float v4f   __attribute__((ext_vector_type(4)));

#define BATCH 32
#define NSEQ  2048
#define DFEAT 512
#define KCB   1024
#define NROWS (BATCH * NSEQ)
#define ROWS_PER_BLOCK 128

// Workspace layout:
//   [0      .. 4096)          float  csq[KCB]        (|c|^2, exact fp32)
//   [4096   .. 4104)          double loss accumulator
//   [4104   .. 4112)          double denominator
//   [8192   .. 8192+1M)       bf16 cbT_hi[KCB][DFEAT]  (hi part, not transposed)
//   [8192+1M .. 8192+2M)      bf16 cbT_lo[KCB][DFEAT]  (lo part, not transposed)
#define WS_LOSS_OFF  4096
#define WS_CBT_OFF   8192
#define CBT_ELEMS    (DFEAT * KCB)

// --------------------------- |c|^2 table (exact fp32) -----------------------
__global__ __launch_bounds__(256) void vq_csq(const float* __restrict__ cb,
                                              float* __restrict__ csq) {
    const int wave = threadIdx.x >> 5;
    const int lane = threadIdx.x & 31;
    const int code = blockIdx.x * 8 + wave;
    const float* __restrict__ row = cb + (long long)code * DFEAT;
    float s = 0.0f;
#pragma unroll
    for (int i = 0; i < DFEAT / 32; ++i) {
        float v = row[lane + i * 32];
        s += v * v;
    }
#pragma unroll
    for (int off = 16; off >= 1; off >>= 1) s += __shfl_xor(s, off, 32);
    if (lane == 0) csq[code] = s;
}

// ------------------- bf16 hi/lo split of codebook ---------------
// cbT[c][d] (not transposed): B-fragment k runs are contiguous per code.
__global__ __launch_bounds__(256) void vq_prep(const float* __restrict__ cb,
                                               bf16* __restrict__ cbT_hi,
                                               bf16* __restrict__ cbT_lo) {
    const int e = blockIdx.x * 256 + threadIdx.x;  // e = c*DFEAT + d
    const int d = e & (DFEAT - 1);
    const int c = e >> 9;                          // DFEAT = 512
    const float f = cb[(long long)c * DFEAT + d];
    const bf16 h = (bf16)f;
    const bf16 l = (bf16)(f - (float)h);
    cbT_hi[e] = h;
    cbT_lo[e] = l;
}

// --------------------------- init loss / denom ------------------------------
__global__ void vq_init(const long long* __restrict__ lens,
                        double* __restrict__ loss,
                        double* __restrict__ denom) {
    if (threadIdx.x == 0) {
        long long s = 0;
        for (int i = 0; i < BATCH; ++i) s += lens[i];
        *loss = 0.0;
        double d = (double)s * (double)DFEAT;
        *denom = (d < 1.0) ? 1.0 : d;
    }
}

// --------------------------- main fused kernel ------------------------------
// 512 blocks x 256 threads (8 waves). Each wave: 16 rows vs all 1024 codes,
// processed as 8 passes of 8 code-tiles (128 codes, 64 acc VGPRs).
__global__ __launch_bounds__(256) void vq_main(
    const float* __restrict__ x, const long long* __restrict__ lens,
    const float* __restrict__ cb, const float* __restrict__ csq,
    const bf16* __restrict__ cbT_hi, const bf16* __restrict__ cbT_lo,
    float* __restrict__ outq, float* __restrict__ outi,
    double* __restrict__ loss) {
    __shared__ int s_idx[ROWS_PER_BLOCK];

    const int tid  = threadIdx.x;
    const int wave = tid >> 5;
    const int lane = tid & 31;
    const int l16  = lane & 15;   // A row / C column
    const int lhi  = lane >> 4;   // half-wave select

    const long long rowBase = (long long)blockIdx.x * ROWS_PER_BLOCK;
    const long long waveRow = rowBase + wave * 16;

    // A fragment source: lane l16 streams row (waveRow + l16).
    const float* __restrict__ xrow = x + (waveRow + l16) * DFEAT;

    float best[8];
    int   bidx[8];
#pragma unroll
    for (int j = 0; j < 8; ++j) { best[j] = 3.402823e38f; bidx[j] = 0; }

    for (int pass = 0; pass < KCB / 128; ++pass) {   // 8 passes
        const int cb0 = pass * 128;
        v8f acc[8] = {};

        for (int d = 0; d < DFEAT; d += 32) {
            // --- build A fragments (ISA 16-bit A layout) ---
            // lane (l16, lhi): elems 0..7  = x[m][d + 8*lhi + 0..7]
            //                  elems 8..15 = x[m][d + 16 + 8*lhi + 0..7]
            const float* p = xrow + d + 8 * lhi;
            v4f q0 = *(const v4f*)(p);
            v4f q1 = *(const v4f*)(p + 4);
            v4f q2 = *(const v4f*)(p + 16);
            v4f q3 = *(const v4f*)(p + 20);
            float fa[16];
#pragma unroll
            for (int e = 0; e < 4; ++e) {
                fa[e]      = q0[e];
                fa[e + 4]  = q1[e];
                fa[e + 8]  = q2[e];
                fa[e + 12] = q3[e];
            }
            v16bf ahi, alo;
#pragma unroll
            for (int e = 0; e < 16; ++e) {
                const float f = fa[e];
                const bf16 h = (bf16)f;
                ahi[e] = h;
                alo[e] = (bf16)(f - (float)h);
            }

            // --- B fragments (lane = code column l16; k in same table order as A) ---
            const long long bo = (long long)(cb0 + l16) * DFEAT + d + 8 * lhi;
            const bf16* __restrict__ bh = cbT_hi + bo;
            const bf16* __restrict__ bl = cbT_lo + bo;

#pragma unroll
            for (int t = 0; t < 8; ++t) {
                v16bf bhi, blo;
#pragma unroll
                for (int s = 0; s < 8; ++s) {
                    bhi[s]     = bh[t * 16 * DFEAT + s];
                    bhi[s + 8] = bh[t * 16 * DFEAT + 16 + s];
                    blo[s]     = bl[t * 16 * DFEAT + s];
                    blo[s + 8] = bl[t * 16 * DFEAT + 16 + s];
                }
                acc[t] = __builtin_amdgcn_wmma_f32_16x16x32_bf16(
                    false, ahi, false, bhi, (short)0, acc[t], false, false);
                acc[t] = __builtin_amdgcn_wmma_f32_16x16x32_bf16(
                    false, ahi, false, blo, (short)0, acc[t], false, false);
                acc[t] = __builtin_amdgcn_wmma_f32_16x16x32_bf16(
                    false, alo, false, bhi, (short)0, acc[t], false, false);
            }
        }

        // score = |c|^2 - 2*dot ; C/D layout: acc[t][j] is row (j + 8*lhi),
        // column (code residue l16).
#pragma unroll
        for (int t = 0; t < 8; ++t) {
            const int code = cb0 + t * 16 + l16;
            const float cs = csq[code];
#pragma unroll
            for (int j = 0; j < 8; ++j) {
                float s = cs - 2.0f * acc[t][j];
                if (s < best[j]) { best[j] = s; bidx[j] = code; }
            }
        }
    }

    // Min+index reduction across the 16 lanes of each half-wave.
#pragma unroll
    for (int off = 1; off < 16; off <<= 1) {
#pragma unroll
        for (int j = 0; j < 8; ++j) {
            float os = __shfl_xor(best[j], off, 32);
            int   oi = __shfl_xor(bidx[j], off, 32);
            if (os < best[j] || (os == best[j] && oi < bidx[j])) {
                best[j] = os;
                bidx[j] = oi;
            }
        }
    }
    if (l16 == 0) {
#pragma unroll
        for (int j = 0; j < 8; ++j) s_idx[wave * 16 + 8 * lhi + j] = bidx[j];
    }
    __syncthreads();

    // ---- output gather + masked select + commitment-loss accumulation ----
    // Block covers 128 consecutive rows -> single batch, single length.
    const int batch = (int)(rowBase / NSEQ);
    const int nBase = (int)(rowBase % NSEQ);
    const long long L = lens[batch];

    float se = 0.0f;
    const long long eBase = rowBase * DFEAT;
#pragma unroll 4
    for (int it = 0; it < (ROWS_PER_BLOCK * DFEAT) / 256; ++it) {
        const int e = it * 256 + tid;
        const int r = e >> 9;            // DFEAT = 512
        const int d = e & (DFEAT - 1);
        const bool m = (long long)(nBase + r) < L;
        const float xv = x[eBase + e];
        const float cv = cb[(long long)s_idx[r] * DFEAT + d];   // exact fp32
        outq[eBase + e] = m ? cv : xv;
        const float diff = cv - xv;
        se += m ? diff * diff : 0.0f;
    }
    if (tid < ROWS_PER_BLOCK) {
        const bool m = (long long)(nBase + tid) < L;
        outi[rowBase + tid] = m ? (float)s_idx[tid] : -1.0f;
    }
#pragma unroll
    for (int off = 16; off >= 1; off >>= 1) se += __shfl_xor(se, off, 32);
    if (lane == 0) atomicAdd(loss, (double)se);
}

// --------------------------- finalize loss ----------------------------------
__global__ void vq_finalize(const double* __restrict__ loss,
                            const double* __restrict__ denom,
                            float* __restrict__ out_loss) {
    if (threadIdx.x == 0 && blockIdx.x == 0)
        *out_loss = (float)(*loss / *denom);
}

// ---------------------------------------------------------------------------
extern "C" void kernel_launch(void* const* d_in, const int* in_sizes, int n_in,
                              void* d_out, int out_size, void* d_ws,
                              size_t ws_size, hipStream_t stream) {
    const float*     x    = (const float*)d_in[0];
    const long long* lens = (const long long*)d_in[1];
    const float*     cb   = (const float*)d_in[2];

    float* outq = (float*)d_out;                       // B*N*D quantized
    float* outi = outq + (long long)NROWS * DFEAT;     // B*N indices (as float)
    float* outl = outi + NROWS;                        // scalar commit loss

    float*  csq    = (float*)d_ws;
    double* loss   = (double*)((char*)d_ws + WS_LOSS_OFF);
    double* denom  = loss + 1;
    bf16*   cbT_hi = (bf16*)((char*)d_ws + WS_CBT_OFF);
    bf16*   cbT_lo = cbT_hi + CBT_ELEMS;

    vq_csq<<<KCB / 8, 256, 0, stream>>>(cb, csq);
    vq_prep<<<CBT_ELEMS / 256, 256, 0, stream>>>(cb, cbT_hi, cbT_lo);
    vq_init<<<1, 32, 0, stream>>>(lens, loss, denom);
    vq_main<<<NROWS / ROWS_PER_BLOCK, 256, 0, stream>>>(
        x, lens, cb, csq, cbT_hi, cbT_lo, outq, outi, loss);
    vq_finalize<<<1, 32, 0, stream>>>(loss, denom, outl);
}
